// MultiLayerGTC_59983513256398
// MI455X (gfx1250) — matrix operand order repaired and checked by reading, not compiled, not run on hardware
//
#include <hip/hip_runtime.h>

// Problem constants (match reference setup_inputs)
#define NN   50000
#define EE   1600000
#define DD   128
#define HH   2
#define CC   64
#define HCC  128
#define DOUT 512          // fused [q|k|v|skip]
#define NTILES 32         // DOUT/16
#define KTILES 4          // DD/32

typedef __attribute__((ext_vector_type(16))) _Float16 v16h;
typedef __attribute__((ext_vector_type(8)))  _Float16 v8h;
typedef __attribute__((ext_vector_type(8)))  float    v8f;

// ---------------------------------------------------------------------------
// Float atomic-max via monotone integer mapping (works for mixed signs).
__device__ __forceinline__ void atomicMaxF32(float* addr, float val) {
    int iv = __float_as_int(val);
    if (iv >= 0) atomicMax((int*)addr, iv);
    else         atomicMin((unsigned int*)addr, (unsigned int)iv);
}

// ---------------------------------------------------------------------------
// Pack one layer's 4 weight matrices [128x128 f32, row-major (k,n)] into the
// per-lane B-fragment order this file uses for v_wmma_f32_16x16x32_f16 (contiguous k, same as A):
// lane L holds column n = ntBase + (L&15); element r2 (0..15) holds
// k = kt*32 + (L>>4)*16 + r2.  Packed as [kt][nt][lane][r2] contiguous f16.
__global__ void pack_weights_kernel(const float* __restrict__ Wq, const float* __restrict__ Wk,
                                    const float* __restrict__ Wv, const float* __restrict__ Ws,
                                    const float* __restrict__ bq, const float* __restrict__ bk,
                                    const float* __restrict__ bv, const float* __restrict__ bs,
                                    _Float16* __restrict__ pB, float* __restrict__ pBias) {
    int idx  = blockIdx.x * blockDim.x + threadIdx.x;     // 0 .. 65535
    int r2   = idx & 15;
    int lane = (idx >> 4) & 31;
    int nt   = (idx >> 9) & 31;
    int kt   = idx >> 14;
    int ng   = nt * 16 + (lane & 15);                     // global output column 0..511
    int hi   = lane >> 4;
    int k    = kt * 32 + hi * 16 + r2;
    int mi   = ng >> 7;                                   // which matrix (q/k/v/s)
    int nc   = ng & 127;
    const float* W = (mi == 0) ? Wq : (mi == 1) ? Wk : (mi == 2) ? Wv : Ws;
    pB[idx] = (_Float16)W[k * HCC + nc];
    if (idx < DOUT) {
        int bmi = idx >> 7;
        const float* b = (bmi == 0) ? bq : (bmi == 1) ? bk : (bmi == 2) ? bv : bs;
        pBias[idx] = b[idx & 127];
    }
}

// ---------------------------------------------------------------------------
__global__ void cvt_f32_to_f16_kernel(const float* __restrict__ x, _Float16* __restrict__ xh, int n) {
    int i = blockIdx.x * blockDim.x + threadIdx.x;
    if (i < n) xh[i] = (_Float16)x[i];
}

// ---------------------------------------------------------------------------
// Fused QKVS GEMM: Y[N,512] = A[N,128](f16) * Wcat(f16) + bias, WMMA f16->f32.
// Each wave owns one 16-row panel and EIGHT 16-col output tiles
// (nt = w, w+4, ..., w+28), so each A fragment load feeds 8 WMMAs.
// blockDim (32,4); grid (N/16).
__global__ __launch_bounds__(128) void gemm_qkvs_kernel(const _Float16* __restrict__ A,
                                                        const _Float16* __restrict__ pB,
                                                        const float* __restrict__ pBias,
                                                        float* __restrict__ Y) {
    int lane    = threadIdx.x;
    int w       = threadIdx.y;                            // 0..3
    int rowBase = blockIdx.x * 16;                        // N = 50000 = 3125*16 exactly
    int m  = lane & 15;
    int hi = lane >> 4;
    const _Float16* arow = A + (size_t)(rowBase + m) * DD;
    v8f acc[8];
#pragma unroll
    for (int j = 0; j < 8; ++j) acc[j] = (v8f){0.f, 0.f, 0.f, 0.f, 0.f, 0.f, 0.f, 0.f};
#pragma unroll
    for (int kt = 0; kt < KTILES; ++kt) {
        // A fragment: lane (m,hi) holds K = kt*32 + hi*16 + [0..16)
        v8h lo  = *(const v8h*)(arow + kt * 32 + hi * 16);
        v8h hi8 = *(const v8h*)(arow + kt * 32 + hi * 16 + 8);
        v16h a;
#pragma unroll
        for (int jj = 0; jj < 8; ++jj) { a[jj] = lo[jj]; a[jj + 8] = hi8[jj]; }
#pragma unroll
        for (int j = 0; j < 8; ++j) {
            int nt = w + 4 * j;
            v16h b = *(const v16h*)(pB + (((size_t)kt * NTILES + nt) * 32 + lane) * 16);
            acc[j] = __builtin_amdgcn_wmma_f32_16x16x32_f16(false, a, false, b, (short)0,
                                                            acc[j], false, false);
        }
    }
#pragma unroll
    for (int j = 0; j < 8; ++j) {
        int nt     = w + 4 * j;
        float bias = pBias[nt * 16 + m];
        float* yb  = Y + (size_t)rowBase * DOUT + nt * 16 + m;
#pragma unroll
        for (int r = 0; r < 8; ++r)                        // C layout: M = r + hi*8, N = m
            yb[(size_t)(r + hi * 8) * DOUT] = acc[j][r] + bias;
    }
}

// ---------------------------------------------------------------------------
__global__ void init_kernel(float* __restrict__ attnOut, float* __restrict__ amax,
                            float* __restrict__ denom) {
    int i = blockIdx.x * blockDim.x + threadIdx.x;
    if (i < NN * HCC) attnOut[i] = 0.0f;
    if (i < NN * HH) { amax[i] = -__builtin_inff(); denom[i] = 0.0f; }
}

// ---------------------------------------------------------------------------
// One wave per edge: alpha[e,h] = (q[dst,h,:] . k[src,h,:]) / 8, atomic segment max.
// Lanes 0-15 -> head 0 (cols 0..63), lanes 16-31 -> head 1 (cols 64..127).
__global__ __launch_bounds__(256) void edge_alpha_kernel(const float* __restrict__ Y,
                                                         const int* __restrict__ src,
                                                         const int* __restrict__ dst,
                                                         float* __restrict__ alpha,
                                                         float* __restrict__ amax) {
    int e = blockIdx.x * blockDim.y + threadIdx.y;
    if (e >= EE) return;
    int lane = threadIdx.x;
    int s = src[e], d = dst[e];
    float4 qv = *(const float4*)(Y + (size_t)d * DOUT + lane * 4);          // q @ col 0
    float4 kv = *(const float4*)(Y + (size_t)s * DOUT + HCC + lane * 4);    // k @ col 128
    float p = qv.x * kv.x + qv.y * kv.y + qv.z * kv.z + qv.w * kv.w;
#pragma unroll
    for (int off = 8; off >= 1; off >>= 1) p += __shfl_xor(p, off, 32);     // per-16-lane-half
    if ((lane & 15) == 0) {
        int h = lane >> 4;
        p *= 0.125f;                                                        // 1/sqrt(64)
        alpha[(size_t)e * HH + h] = p;
        atomicMaxF32(&amax[d * HH + h], p);
    }
}

// ---------------------------------------------------------------------------
__global__ void edge_exp_kernel(float* __restrict__ alphaEx, const int* __restrict__ dst,
                                const float* __restrict__ amax, float* __restrict__ denom) {
    int i = blockIdx.x * blockDim.x + threadIdx.x;
    if (i >= EE * HH) return;
    int e = i >> 1, h = i & 1;
    int d = dst[e];
    float ex = __expf(alphaEx[i] - amax[d * HH + h]);
    alphaEx[i] = ex;
    atomicAdd(&denom[d * HH + h], ex);
}

// ---------------------------------------------------------------------------
// One wave per edge: attnOut[dst,:] += (ex/denom) * v[src,:].
__global__ __launch_bounds__(256) void edge_scatter_kernel(const float* __restrict__ Y,
                                                           const int* __restrict__ src,
                                                           const int* __restrict__ dst,
                                                           const float* __restrict__ ex,
                                                           const float* __restrict__ denom,
                                                           float* __restrict__ attnOut) {
    int e = blockIdx.x * blockDim.y + threadIdx.y;
    if (e >= EE) return;
    int lane = threadIdx.x;
    int s = src[e], d = dst[e];
    int h = lane >> 4;
    float w = ex[(size_t)e * HH + h] / denom[d * HH + h];
    float4 vv = *(const float4*)(Y + (size_t)s * DOUT + 2 * HCC + lane * 4); // v @ col 256
    float* ob = attnOut + (size_t)d * HCC + lane * 4;
    atomicAdd(ob + 0, vv.x * w);
    atomicAdd(ob + 1, vv.y * w);
    atomicAdd(ob + 2, vv.z * w);
    atomicAdd(ob + 3, vv.w * w);
}

// ---------------------------------------------------------------------------
// One wave per node: add skip (Y col 384..511), LayerNorm, write f32 and/or f16.
__global__ __launch_bounds__(256) void layernorm_kernel(const float* __restrict__ attnOut,
                                                        const float* __restrict__ Y,
                                                        const float* __restrict__ g,
                                                        const float* __restrict__ be,
                                                        float* __restrict__ outF,
                                                        _Float16* __restrict__ outH,
                                                        int flags) {
    int n = blockIdx.x * blockDim.y + threadIdx.y;
    if (n >= NN) return;
    int lane = threadIdx.x;
    float v[4];
    float s = 0.0f, ss = 0.0f;
#pragma unroll
    for (int j = 0; j < 4; ++j) {
        int c = lane * 4 + j;
        float t = attnOut[(size_t)n * HCC + c] + Y[(size_t)n * DOUT + 3 * HCC + c];
        v[j] = t; s += t; ss += t * t;
    }
#pragma unroll
    for (int off = 16; off >= 1; off >>= 1) {
        s  += __shfl_xor(s,  off, 32);
        ss += __shfl_xor(ss, off, 32);
    }
    float mu   = s * (1.0f / HCC);
    float var  = ss * (1.0f / HCC) - mu * mu;
    float rstd = rsqrtf(var + 1e-5f);
#pragma unroll
    for (int j = 0; j < 4; ++j) {
        int c = lane * 4 + j;
        float o = g[c] * (v[j] - mu) * rstd + be[c];
        if (flags & 1) outF[(size_t)n * HCC + c] = o;
        if (flags & 2) outH[(size_t)n * HCC + c] = (_Float16)o;
    }
}

// ---------------------------------------------------------------------------
extern "C" void kernel_launch(void* const* d_in, const int* in_sizes, int n_in,
                              void* d_out, int out_size, void* d_ws, size_t ws_size,
                              hipStream_t stream) {
    (void)in_sizes; (void)n_in; (void)out_size; (void)ws_size;

    const float* x      = (const float*)d_in[0];
    const int*   ei     = (const int*)d_in[1];
    const int*   srcIdx = ei;            // edge_index[0]
    const int*   dstIdx = ei + EE;       // edge_index[1]

    // Workspace carve-out (256B aligned), ~155 MB total.
    char* ws = (char*)d_ws;
    size_t off = 0;
    auto wsalloc = [&](size_t bytes) -> void* {
        void* p = ws + off;
        off += (bytes + 255) & ~(size_t)255;
        return p;
    };
    _Float16* xh      = (_Float16*)wsalloc((size_t)NN * DD * sizeof(_Float16));
    float*    Y       = (float*)   wsalloc((size_t)NN * DOUT * sizeof(float));
    float*    attnOut = (float*)   wsalloc((size_t)NN * HCC * sizeof(float));
    float*    alphaEx = (float*)   wsalloc((size_t)EE * HH * sizeof(float));
    float*    amax    = (float*)   wsalloc((size_t)NN * HH * sizeof(float));
    float*    denom   = (float*)   wsalloc((size_t)NN * HH * sizeof(float));
    _Float16* pB0     = (_Float16*)wsalloc(65536 * sizeof(_Float16));
    float*    pBias0  = (float*)   wsalloc(DOUT * sizeof(float));
    _Float16* pB1     = (_Float16*)wsalloc(65536 * sizeof(_Float16));
    float*    pBias1  = (float*)   wsalloc(DOUT * sizeof(float));

    // Weight packing (both layers) + input conversion.
    pack_weights_kernel<<<65536 / 256, 256, 0, stream>>>(
        (const float*)d_in[2], (const float*)d_in[3], (const float*)d_in[4], (const float*)d_in[5],
        (const float*)d_in[6], (const float*)d_in[7], (const float*)d_in[8], (const float*)d_in[9],
        pB0, pBias0);
    pack_weights_kernel<<<65536 / 256, 256, 0, stream>>>(
        (const float*)d_in[12], (const float*)d_in[13], (const float*)d_in[14], (const float*)d_in[15],
        (const float*)d_in[16], (const float*)d_in[17], (const float*)d_in[18], (const float*)d_in[19],
        pB1, pBias1);
    cvt_f32_to_f16_kernel<<<(NN * DD + 255) / 256, 256, 0, stream>>>(x, xh, NN * DD);

    for (int li = 0; li < 2; ++li) {
        _Float16* pB    = (li == 0) ? pB0 : pB1;
        float*    pBias = (li == 0) ? pBias0 : pBias1;
        const float* g  = (const float*)d_in[li == 0 ? 10 : 20];
        const float* be = (const float*)d_in[li == 0 ? 11 : 21];

        gemm_qkvs_kernel<<<dim3(NN / 16), dim3(32, 4), 0, stream>>>(xh, pB, pBias, Y);
        init_kernel<<<(NN * HCC + 255) / 256, 256, 0, stream>>>(attnOut, amax, denom);
        edge_alpha_kernel<<<dim3((EE + 7) / 8), dim3(32, 8), 0, stream>>>(Y, srcIdx, dstIdx, alphaEx, amax);
        edge_exp_kernel<<<(EE * HH + 255) / 256, 256, 0, stream>>>(alphaEx, dstIdx, amax, denom);
        edge_scatter_kernel<<<dim3((EE + 7) / 8), dim3(32, 8), 0, stream>>>(Y, srcIdx, dstIdx,
                                                                            alphaEx, denom, attnOut);
        // Layer 0: write f16 features for next GEMM. Layer 1: write final f32 output.
        layernorm_kernel<<<dim3((NN + 7) / 8), dim3(32, 8), 0, stream>>>(
            attnOut, Y, g, be,
            (li == 0) ? (float*)nullptr : (float*)d_out,
            xh,
            (li == 0) ? 2 : 1);
    }
}
